// MXFP4Linear_46591805227029
// MI455X (gfx1250) — matrix operand order repaired and checked by reading, not compiled, not run on hardware
//
#include <hip/hip_runtime.h>

typedef __attribute__((ext_vector_type(16))) _Float16 v16h;
typedef __attribute__((ext_vector_type(8)))  _Float16 v8h;
typedef __attribute__((ext_vector_type(8)))  float    v8f;
typedef __attribute__((ext_vector_type(4)))  int      v4i;

#define IN_DIM   4096
#define OUT_DIM  4096
#define BM       128
#define BN       128
#define BK       32     // fused-path K step
#define LDSTR    40     // fused-path LDS row stride (halves)
#define GBK      64     // fp16 GEMM K step (2 WMMA k-slices)
#define GSTR     72     // fp16 GEMM LDS row stride (halves): 144B, 16B-aligned, conflict-free

#define AS1 __attribute__((address_space(1)))
#define AS3 __attribute__((address_space(3)))

// ---- gfx1250 async-to-LDS path (guarded so absence degrades, not breaks) ----
#if defined(__HIP_DEVICE_COMPILE__) && __has_builtin(__builtin_amdgcn_global_load_async_to_lds_b128)
#define HAVE_ASYNC 1
#else
#define HAVE_ASYNC 0
#endif

#if defined(__HIP_DEVICE_COMPILE__) && __has_builtin(__builtin_amdgcn_s_wait_asynccnt)
#define WAIT_ASYNC(n) __builtin_amdgcn_s_wait_asynccnt(n)
#elif HAVE_ASYNC
#define WAIT_ASYNC(n) asm volatile("s_wait_asynccnt %0" :: "i"(n) : "memory")
#else
#define WAIT_ASYNC(n) do {} while (0)
#endif

__device__ __forceinline__ void copy16_to_lds(const _Float16* g, _Float16* l) {
#if HAVE_ASYNC
    __builtin_amdgcn_global_load_async_to_lds_b128((AS1 v4i*)g, (AS3 v4i*)l, 0, 0);
#else
    *reinterpret_cast<int4*>(l) = *reinterpret_cast<const int4*>(g);
#endif
}

// Branch-free FP4 (E2M1) -> fp16 decode, then scale.
// nib[3] = sign, nib[2:0] -> {0, 0.5, 1, 1.5, 2, 3, 4, 6}
__device__ __forceinline__ _Float16 fp4_dq(unsigned nib, float scale) {
    const unsigned long long lutLo = 0x3E003C0038000000ULL; // 1.5,1.0,0.5,0.0
    const unsigned long long lutHi = 0x4600440042004000ULL; // 6.0,4.0,3.0,2.0
    unsigned idx = nib & 7u;
    unsigned long long sel = (idx & 4u) ? lutHi : lutLo;
    unsigned short bits = (unsigned short)((sel >> ((idx & 3u) * 16u)) & 0xFFFFu);
    bits |= (unsigned short)((nib & 8u) << 12);
    _Float16 h;
    __builtin_memcpy(&h, &bits, sizeof(h));
    return (_Float16)((float)h * scale);
}

// =============================== prepass kernels =============================

// x fp32 -> fp16, 8 elements/thread
__global__ __launch_bounds__(256)
void cvt_x_kernel(const float* __restrict__ x, _Float16* __restrict__ xh) {
    size_t i = ((size_t)blockIdx.x * 256 + threadIdx.x) * 8;
    const float4 a = *reinterpret_cast<const float4*>(x + i);
    const float4 b = *reinterpret_cast<const float4*>(x + i + 4);
    v8h h;
    h[0] = (_Float16)a.x; h[1] = (_Float16)a.y; h[2] = (_Float16)a.z; h[3] = (_Float16)a.w;
    h[4] = (_Float16)b.x; h[5] = (_Float16)b.y; h[6] = (_Float16)b.z; h[7] = (_Float16)b.w;
    *reinterpret_cast<v8h*>(xh + i) = h;
}

// W fp4 -> fp16 once (removes 64x redundant dequant from the GEMM loop).
// Thread g: int4 = 4 packed bytes = 8 weights, all inside one 32-elem scale block.
__global__ __launch_bounds__(256)
void dq_w_kernel(const int* __restrict__ qw, const float* __restrict__ scales,
                 _Float16* __restrict__ wh) {
    size_t g = (size_t)blockIdx.x * 256 + threadIdx.x;
    const int4 pk = *reinterpret_cast<const int4*>(qw + g * 4);
    const float s = scales[g >> 2];
    int b[4] = {pk.x, pk.y, pk.z, pk.w};
    v8h h;
#pragma unroll
    for (int j = 0; j < 4; ++j) {
        unsigned byte = (unsigned)b[j] & 0xFFu;
        h[2 * j + 0] = fp4_dq((byte >> 4) & 15u, s);  // high nibble first
        h[2 * j + 1] = fp4_dq(byte & 15u, s);
    }
    *reinterpret_cast<v8h*>(wh + g * 8) = h;
}

// ========================= fp16 GEMM (async double-buffer) ===================
// out[M,N] = xh[M,K] * wh[N,K]^T + bias ; both operands L2-resident fp16.
__global__ __launch_bounds__(256)
void gemm_fp16_wmma(const _Float16* __restrict__ xh,
                    const _Float16* __restrict__ wh,
                    const float* __restrict__ bias,
                    float* __restrict__ out) {
    __shared__ _Float16 sa[2][BM * GSTR];
    __shared__ _Float16 sb[2][BN * GSTR];

    const int tid  = threadIdx.x;
    const int lane = tid & 31;
    const int wave = tid >> 5;
    const int hl   = lane >> 4;
    const int r    = lane & 15;
    const int wm   = wave & 1;    // 2 wave rows  x 64 M
    const int wn   = wave >> 1;   // 4 wave cols  x 32 N

    const int n0 = blockIdx.x * BN;
    const int m0 = blockIdx.y * BM;

    v8f acc[4][2];
#pragma unroll
    for (int mi = 0; mi < 4; ++mi)
#pragma unroll
        for (int ni = 0; ni < 2; ++ni)
            acc[mi][ni] = (v8f){0.f, 0.f, 0.f, 0.f, 0.f, 0.f, 0.f, 0.f};

    // 8 async b128 copies per thread per stage (4 for A, 4 for B)
    auto stage = [&](int buf, int kt) {
#pragma unroll
        for (int i = 0; i < 4; ++i) {
            int c   = tid + i * 256;   // 1024 b128 chunks per operand
            int row = c >> 3;          // 128B (8x16B) per 64-half row
            int c8  = c & 7;
            copy16_to_lds(xh + (size_t)(m0 + row) * IN_DIM + kt + c8 * 8,
                          &sa[buf][row * GSTR + c8 * 8]);
            copy16_to_lds(wh + (size_t)(n0 + row) * IN_DIM + kt + c8 * 8,
                          &sb[buf][row * GSTR + c8 * 8]);
        }
    };

    const int NK = IN_DIM / GBK;   // 64 iterations
    int buf = 0;
    stage(0, 0);

    for (int it = 0; it < NK; ++it) {
        if (it + 1 < NK) {
            stage(buf ^ 1, (it + 1) * GBK);  // prefetch next stage (async)
            WAIT_ASYNC(8);                   // current stage's 8 copies done
        } else {
            WAIT_ASYNC(0);
        }
        __syncthreads();

#pragma unroll
        for (int ks = 0; ks < 2; ++ks) {
            const int ko = ks * 32;

            v16h afrag[4];
#pragma unroll
            for (int mi = 0; mi < 4; ++mi) {
                int arow = wm * 64 + mi * 16 + r;
                v8h alo = *reinterpret_cast<const v8h*>(&sa[buf][arow * GSTR + ko + hl * 8]);
                v8h ahi = *reinterpret_cast<const v8h*>(&sa[buf][arow * GSTR + ko + 16 + hl * 8]);
                v16h a;
#pragma unroll
                for (int e = 0; e < 8; ++e) { a[e] = alo[e]; a[e + 8] = ahi[e]; }
                afrag[mi] = a;
            }

            v16h bfrag[2];
#pragma unroll
            for (int ni = 0; ni < 2; ++ni) {
                int bcol = wn * 32 + ni * 16 + r;
                v8h blo = *reinterpret_cast<const v8h*>(&sb[buf][bcol * GSTR + ko + hl * 8]);
                v8h bhi = *reinterpret_cast<const v8h*>(&sb[buf][bcol * GSTR + ko + 16 + hl * 8]);
                v16h b;
#pragma unroll
                for (int e = 0; e < 8; ++e) { b[e] = blo[e]; b[e + 8] = bhi[e]; }
                bfrag[ni] = b;
            }

#pragma unroll
            for (int mi = 0; mi < 4; ++mi)
#pragma unroll
                for (int ni = 0; ni < 2; ++ni)
                    acc[mi][ni] = __builtin_amdgcn_wmma_f32_16x16x32_f16(
                        false, afrag[mi], false, bfrag[ni],
                        (short)0, acc[mi][ni], false, false);
        }

        __syncthreads();
        buf ^= 1;
    }

#pragma unroll
    for (int ni = 0; ni < 2; ++ni) {
        const int col = n0 + wn * 32 + ni * 16 + r;
        const float bz = bias[col];
#pragma unroll
        for (int mi = 0; mi < 4; ++mi) {
            const int mbase = m0 + wm * 64 + mi * 16 + hl * 8;
#pragma unroll
            for (int reg = 0; reg < 8; ++reg)
                out[(size_t)(mbase + reg) * OUT_DIM + col] = acc[mi][ni][reg] + bz;
        }
    }
}

// ======================= fused fallback (small workspace) ====================
__global__ __launch_bounds__(256)
void mxfp4_linear_wmma(const float* __restrict__ x,
                       const int*   __restrict__ qw,
                       const float* __restrict__ scales,
                       const float* __restrict__ bias,
                       float*       __restrict__ out) {
    __shared__ _Float16 lds_a[BM * LDSTR];
    __shared__ _Float16 lds_b[BN * LDSTR];

    const int tid  = threadIdx.x;
    const int lane = tid & 31;
    const int wave = tid >> 5;
    const int hl   = lane >> 4;
    const int r    = lane & 15;
    const int wm   = wave & 1;
    const int wn   = wave >> 1;

    const int n0 = blockIdx.x * BN;
    const int m0 = blockIdx.y * BM;

    v8f acc[4][2];
#pragma unroll
    for (int mi = 0; mi < 4; ++mi)
#pragma unroll
        for (int ni = 0; ni < 2; ++ni)
            acc[mi][ni] = (v8f){0.f, 0.f, 0.f, 0.f, 0.f, 0.f, 0.f, 0.f};

    for (int kt32 = 0; kt32 < IN_DIM / BK; ++kt32) {
        const int kt = kt32 * BK;
#pragma unroll
        for (int i = 0; i < 4; ++i) {
            int f = tid + i * 256;
            int row = f >> 3, c4 = f & 7;
            const float4 v = *reinterpret_cast<const float4*>(
                x + (size_t)(m0 + row) * IN_DIM + kt + c4 * 4);
            _Float16* dst = &lds_a[row * LDSTR + c4 * 4];
            dst[0] = (_Float16)v.x; dst[1] = (_Float16)v.y;
            dst[2] = (_Float16)v.z; dst[3] = (_Float16)v.w;
        }
#pragma unroll
        for (int i = 0; i < 2; ++i) {
            int g = tid + i * 256;
            int row = g >> 2, q4 = g & 3;
            const int4 pk = *reinterpret_cast<const int4*>(
                qw + (size_t)(n0 + row) * (IN_DIM / 2) + (kt >> 1) + q4 * 4);
            const float s = scales[(size_t)(n0 + row) * (IN_DIM / 32) + kt32];
            _Float16* dst = &lds_b[row * LDSTR + q4 * 8];
            int b[4] = {pk.x, pk.y, pk.z, pk.w};
#pragma unroll
            for (int j = 0; j < 4; ++j) {
                unsigned byte = (unsigned)b[j] & 0xFFu;
                dst[2 * j + 0] = fp4_dq((byte >> 4) & 15u, s);
                dst[2 * j + 1] = fp4_dq(byte & 15u, s);
            }
        }
        __syncthreads();

        v16h afrag[4];
#pragma unroll
        for (int mi = 0; mi < 4; ++mi) {
            int arow = wm * 64 + mi * 16 + r;
            v8h alo = *reinterpret_cast<const v8h*>(&lds_a[arow * LDSTR + hl * 8]);
            v8h ahi = *reinterpret_cast<const v8h*>(&lds_a[arow * LDSTR + 16 + hl * 8]);
            v16h a;
#pragma unroll
            for (int e = 0; e < 8; ++e) { a[e] = alo[e]; a[e + 8] = ahi[e]; }
            afrag[mi] = a;
        }
        v16h bfrag[2];
#pragma unroll
        for (int ni = 0; ni < 2; ++ni) {
            int bcol = wn * 32 + ni * 16 + r;
            v8h blo = *reinterpret_cast<const v8h*>(&lds_b[bcol * LDSTR + hl * 8]);
            v8h bhi = *reinterpret_cast<const v8h*>(&lds_b[bcol * LDSTR + 16 + hl * 8]);
            v16h b;
#pragma unroll
            for (int e = 0; e < 8; ++e) { b[e] = blo[e]; b[e + 8] = bhi[e]; }
            bfrag[ni] = b;
        }
#pragma unroll
        for (int mi = 0; mi < 4; ++mi)
#pragma unroll
            for (int ni = 0; ni < 2; ++ni)
                acc[mi][ni] = __builtin_amdgcn_wmma_f32_16x16x32_f16(
                    false, afrag[mi], false, bfrag[ni],
                    (short)0, acc[mi][ni], false, false);
        __syncthreads();
    }

#pragma unroll
    for (int ni = 0; ni < 2; ++ni) {
        const int col = n0 + wn * 32 + ni * 16 + r;
        const float bz = bias[col];
#pragma unroll
        for (int mi = 0; mi < 4; ++mi) {
            const int mbase = m0 + wm * 64 + mi * 16 + hl * 8;
#pragma unroll
            for (int reg = 0; reg < 8; ++reg)
                out[(size_t)(mbase + reg) * OUT_DIM + col] = acc[mi][ni][reg] + bz;
        }
    }
}

// ================================== launch ===================================
extern "C" void kernel_launch(void* const* d_in, const int* in_sizes, int n_in,
                              void* d_out, int out_size, void* d_ws, size_t ws_size,
                              hipStream_t stream) {
    const float* x      = (const float*)d_in[0];
    const int*   qw     = (const int*)d_in[1];
    const float* scales = (const float*)d_in[2];
    const float* bias   = (const float*)d_in[3];
    float*       out    = (float*)d_out;

    const int M = in_sizes[0] / IN_DIM;             // 8192
    const size_t xBytes = (size_t)M * IN_DIM * sizeof(_Float16);        // 64 MiB
    const size_t wBytes = (size_t)OUT_DIM * IN_DIM * sizeof(_Float16);  // 32 MiB

    if (ws_size >= xBytes + wBytes) {
        _Float16* xh = (_Float16*)d_ws;
        _Float16* wh = (_Float16*)((char*)d_ws + xBytes);
        // prepass: one-time convert/dequant (both results fit in 192MB L2 together)
        cvt_x_kernel<<<(unsigned)((size_t)M * IN_DIM / (256 * 8)), 256, 0, stream>>>(x, xh);
        dq_w_kernel<<<(unsigned)((size_t)OUT_DIM * IN_DIM / (256 * 8)), 256, 0, stream>>>(qw, scales, wh);
        gemm_fp16_wmma<<<dim3(OUT_DIM / BN, M / BM), 256, 0, stream>>>(xh, wh, bias, out);
    } else {
        mxfp4_linear_wmma<<<dim3(OUT_DIM / BN, M / BM), 256, 0, stream>>>(x, qw, scales, bias, out);
    }
}
